// SelfAttention_38835094291005
// MI455X (gfx1250) — matrix operand order repaired and checked by reading, not compiled, not run on hardware
//
#include <hip/hip_runtime.h>

// Problem constants (from reference)
#define B_  2
#define N_  2048
#define D_  2048
#define H_  16
#define HD_ 128
#define M_  (B_ * N_)   // 4096 rows

typedef __bf16 bf16_t;
typedef __attribute__((ext_vector_type(16))) __bf16 v16bf;
typedef __attribute__((ext_vector_type(8)))  __bf16 v8bf;
typedef __attribute__((ext_vector_type(8)))  float  v8f;

__device__ __forceinline__ bf16_t f2bf(float f) {
    unsigned u = __builtin_bit_cast(unsigned, f);
    unsigned r = u + 0x7FFFu + ((u >> 16) & 1u);   // round-to-nearest-even
    unsigned short h = (unsigned short)(r >> 16);
    return __builtin_bit_cast(bf16_t, h);
}
// Pack two floats into one dword of 2x bf16 (pure 32-bit VALU)
__device__ __forceinline__ unsigned pk2bf(float lo, float hi) {
    unsigned ul = __builtin_bit_cast(unsigned, lo);
    unsigned uh = __builtin_bit_cast(unsigned, hi);
    ul = (ul + 0x7FFFu + ((ul >> 16) & 1u)) >> 16;
    uh = (uh + 0x7FFFu + ((uh >> 16) & 1u)) & 0xFFFF0000u;
    return ul | uh;
}
__device__ __forceinline__ v16bf cat16(v8bf lo, v8bf hi) {
    return __builtin_shufflevector(lo, hi, 0,1,2,3,4,5,6,7,8,9,10,11,12,13,14,15);
}
__device__ __forceinline__ v8f wmma_bf16(v16bf a, v16bf b, v8f c) {
    return __builtin_amdgcn_wmma_f32_16x16x32_bf16(false, a, false, b, (short)0, c,
                                                   false, false);
}

// ------------------------- DPP16 cross-lane helpers -------------------------
template <int CTRL>
__device__ __forceinline__ float dpp_f32(float x) {
    return __builtin_bit_cast(float,
        __builtin_amdgcn_mov_dpp(__builtin_bit_cast(int, x), CTRL, 0xF, 0xF, true));
}
#define DPP_XOR1        0xB1   // quad_perm [1,0,3,2]
#define DPP_XOR2        0x4E   // quad_perm [2,3,0,1]
#define DPP_HALF_MIRROR 0x141
#define DPP_MIRROR      0x140
__device__ __forceinline__ float red_max16(float x) {
    x = fmaxf(x, dpp_f32<DPP_XOR1>(x));
    x = fmaxf(x, dpp_f32<DPP_XOR2>(x));
    x = fmaxf(x, dpp_f32<DPP_HALF_MIRROR>(x));
    x = fmaxf(x, dpp_f32<DPP_MIRROR>(x));
    return x;
}
__device__ __forceinline__ float red_sum16(float x) {
    x += dpp_f32<DPP_XOR1>(x);
    x += dpp_f32<DPP_XOR2>(x);
    x += dpp_f32<DPP_HALF_MIRROR>(x);
    x += dpp_f32<DPP_MIRROR>(x);
    return x;
}

// ------------------------- WMMA fragment loads ------------------------------
// A operand (16x32): lanes 0-15 hold K {0..7,16..23}, lanes 16-31 K {8..15,24..31}
__device__ __forceinline__ v16bf load_a_frag(const bf16_t* base, int row, int ldk,
                                             int kofs, int l16, int lh) {
    const bf16_t* p = base + (row + l16) * ldk + kofs + lh * 8;
    return cat16(*(const v8bf*)p, *(const v8bf*)(p + 16));
}
// B operand (32x16) from LDS [col][k]: lanes 0-15 hold K {0..7,16..23}, lanes 16-31 K {8..15,24..31}
__device__ __forceinline__ v16bf load_b_frag(const bf16_t* base, int col, int ldk,
                                             int kofs, int l16, int lh) {
    const bf16_t* p = base + (col + l16) * ldk + kofs + lh * 8;
    return cat16(*(const v8bf*)p, *(const v8bf*)(p + 16));
}

// ------------------------- Tensor Data Mover --------------------------------
#if __has_builtin(__builtin_amdgcn_tensor_load_to_lds) && \
    __has_builtin(__builtin_amdgcn_s_wait_tensorcnt)
#define HAVE_TDM 1
#else
#define HAVE_TDM 0
#endif

#if HAVE_TDM
typedef __attribute__((ext_vector_type(4))) unsigned tdm_v4u;
typedef __attribute__((ext_vector_type(8))) int      tdm_v8i;
typedef __attribute__((ext_vector_type(4))) int      tdm_v4i;

// 2D bf16 tile load: tile_d0 elems per row, tile_d1 rows, source row stride
// `stride` elems; LDS rows padded by (pad_amount+1) dwords after each
// 2^(pad_interval+1) dwords (set to exactly one tile row).
__device__ __forceinline__ void tdm_load_2d(void* lds_dst, const void* gsrc,
                                            unsigned tile_d0, unsigned tile_d1,
                                            unsigned stride,
                                            unsigned pad_interval, unsigned pad_amount)
{
    unsigned long long ga = (unsigned long long)(uintptr_t)gsrc;
    unsigned lds = (unsigned)(uintptr_t)lds_dst;
    tdm_v4u g0;
    g0[0] = 1u;                                             // count=1
    g0[1] = lds;                                            // lds_addr
    g0[2] = (unsigned)(ga & 0xFFFFFFFFu);                   // global_addr lo
    g0[3] = (unsigned)((ga >> 32) & 0x01FFFFFFu) | (2u << 30);  // addr hi | type=2
    tdm_v8i g1;
    g1[0] = (int)((1u << 16) | (1u << 20) |                 // data_size=2B, pad_enable
                  (pad_interval << 22) | (pad_amount << 25));
    g1[1] = (int)(tile_d0 << 16);                           // tensor_dim0 = tile_d0
    g1[2] = (int)(((tile_d0 >> 16) & 0xFFFFu) | (tile_d1 << 16)); // tensor_dim1
    g1[3] = (int)(((tile_d1 >> 16) & 0xFFFFu) | (tile_d0 << 16)); // tile_dim0
    g1[4] = (int)(tile_d1 & 0xFFFFu);                       // tile_dim1 (tile_dim2=0)
    g1[5] = (int)stride;                                    // tensor_dim0_stride lo
    g1[6] = 0;
    g1[7] = 0;
    tdm_v4i gz = {0, 0, 0, 0};
#if defined(__clang_major__) && (__clang_major__ >= 23)
    tdm_v8i gz8 = {0, 0, 0, 0, 0, 0, 0, 0};
    __builtin_amdgcn_tensor_load_to_lds(g0, g1, gz, gz, gz8, 0);
#else
    __builtin_amdgcn_tensor_load_to_lds(g0, g1, gz, gz, 0);
#endif
}
__device__ __forceinline__ void tdm_wait() { __builtin_amdgcn_s_wait_tensorcnt(0); }
#endif

// Manual bf16 tile copy fallback
__device__ __forceinline__ void fill_tile_128x64(bf16_t* dst, const bf16_t* src,
                                                 int stride, int tid) {
    #pragma unroll
    for (int i = 0; i < 4; ++i) {
        int idx = tid + i * 256;          // 1024 chunks of 8 elems
        int r = idx >> 3;
        int c = (idx & 7) * 8;
        *(v8bf*)(dst + r * 72 + c) = *(const v8bf*)(src + (size_t)r * stride + c);
    }
}
__device__ __forceinline__ void fill_tile_64x128(bf16_t* dst, const bf16_t* src,
                                                 int stride, int tid) {
    #pragma unroll
    for (int i = 0; i < 4; ++i) {
        int idx = tid + i * 256;          // 1024 chunks of 8 elems
        int r = idx >> 4;
        int c = (idx & 15) * 8;
        *(v8bf*)(dst + r * 136 + c) = *(const v8bf*)(src + (size_t)r * stride + c);
    }
}

// ---------------------------------------------------------------------------
// Pre-pass 1: x (fp32) -> bf16, flat
__global__ __launch_bounds__(256)
void cvt_x_kernel(const float* __restrict__ src, bf16_t* __restrict__ dst)
{
    size_t i = ((size_t)blockIdx.x * 256 + threadIdx.x) * 8;
    float4 a = *(const float4*)(src + i);
    float4 b = *(const float4*)(src + i + 4);
    uint4 p = make_uint4(pk2bf(a.x, a.y), pk2bf(a.z, a.w),
                         pk2bf(b.x, b.y), pk2bf(b.z, b.w));
    *(uint4*)(dst + i) = p;
}

// Pre-pass 2: W[k][n] (fp32) -> Wt[n][k] (bf16), 64x64 tiles through LDS
__global__ __launch_bounds__(256)
void wtrans_kernel(const float* __restrict__ W, bf16_t* __restrict__ Wt)
{
    __shared__ bf16_t T[64 * 72];
    const int tid = threadIdx.x;
    const int k0 = blockIdx.y * 64;
    const int n0 = blockIdx.x * 64;
    #pragma unroll
    for (int i = 0; i < 4; ++i) {
        int idx = tid + i * 256;          // 0..1023
        int r   = idx >> 4;               // 64 rows, 16 float4/row
        int c4  = (idx & 15) * 4;
        float4 v = *(const float4*)(W + (size_t)(k0 + r) * D_ + n0 + c4);
        T[(c4 + 0) * 72 + r] = f2bf(v.x);
        T[(c4 + 1) * 72 + r] = f2bf(v.y);
        T[(c4 + 2) * 72 + r] = f2bf(v.z);
        T[(c4 + 3) * 72 + r] = f2bf(v.w);
    }
    __syncthreads();
    #pragma unroll
    for (int i = 0; i < 2; ++i) {
        int idx = tid + i * 256;          // 0..511
        int c   = idx >> 3;
        int kk  = (idx & 7) * 8;
        *(v8bf*)(Wt + (size_t)(n0 + c) * D_ + k0 + kk) = *(const v8bf*)(T + c * 72 + kk);
    }
}

// ---------------------------------------------------------------------------
// GEMM: C[M,D] = A[M,K] * Wt[n][k]^T + bias (A, Wt both bf16, TDM double-buffered)
// ---------------------------------------------------------------------------
#define MODE_V 0
#define MODE_K 1
#define MODE_Q 2
#define MODE_OUT 3

template <int MODE>
__global__ __launch_bounds__(256)
void gemm_kernel(const bf16_t* __restrict__ A, const bf16_t* __restrict__ Wt,
                 const float* __restrict__ bias,
                 const float* __restrict__ cosb, const float* __restrict__ sinb,
                 bf16_t* __restrict__ out_bf, float* __restrict__ out_f)
{
    constexpr int BM = 128, BN = 128, BK = 64;
    constexpr int LDT = BK + 8;           // 144B rows: 16B aligned, conflict-free

    __shared__ bf16_t As[2][BM * LDT];
    __shared__ bf16_t Bs[2][BN * LDT];

    const int tid  = threadIdx.x;
    const int wave = tid >> 5;
    const int lane = tid & 31;
    const int l16  = lane & 15;
    const int lh   = lane >> 4;
    const int wm   = wave & 3;            // 4 waves along M
    const int wn   = wave >> 2;           // 2 waves along N
    const int bm0  = blockIdx.y * BM;
    const int bn0  = blockIdx.x * BN;

    const bf16_t* Ab = A  + (size_t)bm0 * D_;
    const bf16_t* Bb = Wt + (size_t)bn0 * D_;

    v8f acc[2][4];
    const v8f vzero = {0.f,0.f,0.f,0.f,0.f,0.f,0.f,0.f};
    #pragma unroll
    for (int i = 0; i < 2; ++i)
        #pragma unroll
        for (int j = 0; j < 4; ++j) acc[i][j] = vzero;

#if HAVE_TDM
    if (wave == 0) {
        tdm_load_2d(&As[0][0], Ab, BK, BM, D_, 4, 3);   // 128B rows + 16B pad
        tdm_load_2d(&Bs[0][0], Bb, BK, BN, D_, 4, 3);
    }
#else
    fill_tile_128x64(&As[0][0], Ab, D_, tid);
    fill_tile_128x64(&Bs[0][0], Bb, D_, tid);
#endif

    int cur = 0;
    for (int k0 = 0; k0 < D_; k0 += BK) {
#if HAVE_TDM
        if (wave == 0) tdm_wait();
#endif
        __syncthreads();
        if (k0 + BK < D_) {
#if HAVE_TDM
            if (wave == 0) {
                tdm_load_2d(&As[cur ^ 1][0], Ab + k0 + BK, BK, BM, D_, 4, 3);
                tdm_load_2d(&Bs[cur ^ 1][0], Bb + k0 + BK, BK, BN, D_, 4, 3);
            }
#else
            fill_tile_128x64(&As[cur ^ 1][0], Ab + k0 + BK, D_, tid);
            fill_tile_128x64(&Bs[cur ^ 1][0], Bb + k0 + BK, D_, tid);
#endif
        }
        const bf16_t* Asc = &As[cur][0];
        const bf16_t* Bsc = &Bs[cur][0];
        #pragma unroll
        for (int ks = 0; ks < BK; ks += 32) {
            v16bf af[2], bfv[4];
            #pragma unroll
            for (int tm = 0; tm < 2; ++tm)
                af[tm] = load_a_frag(Asc, wm * 32 + tm * 16, LDT, ks, l16, lh);
            #pragma unroll
            for (int tn = 0; tn < 4; ++tn)
                bfv[tn] = load_b_frag(Bsc, wn * 64 + tn * 16, LDT, ks, l16, lh);
            #pragma unroll
            for (int tm = 0; tm < 2; ++tm)
                #pragma unroll
                for (int tn = 0; tn < 4; ++tn)
                    acc[tm][tn] = wmma_bf16(af[tm], bfv[tn], acc[tm][tn]);
        }
        cur ^= 1;
    }

    // Epilogue. C/D layout: VGPR j -> row j + 8*lh, col = l16 within 16x16 tile.
    #pragma unroll
    for (int tm = 0; tm < 2; ++tm) {
        #pragma unroll
        for (int tn = 0; tn < 4; ++tn) {
            const int col = bn0 + wn * 64 + tn * 16 + l16;
            const float bv = bias[col];
            #pragma unroll
            for (int j = 0; j < 8; ++j) {
                const int mrow = bm0 + wm * 32 + tm * 16 + j + lh * 8;
                float v = acc[tm][tn][j] + bv;
                if (MODE == MODE_OUT) {
                    out_f[(size_t)mrow * D_ + col] = v;
                } else {
                    const int b  = mrow >> 11;
                    const int ns = mrow & (N_ - 1);
                    const int h  = col >> 7;
                    const int dh = col & (HD_ - 1);
                    if (MODE != MODE_V) {
                        const float c = cosb[ns * HD_ + dh];
                        const float s = sinb[ns * HD_ + dh];
                        const float p = dpp_f32<DPP_XOR1>(v);   // partner col^1
                        v = (dh & 1) ? (v * c + p * s) : (v * c - p * s);
                        if (MODE == MODE_Q) v *= (1.0f / (float)HD_);  // scale^2
                    }
                    out_bf[(((size_t)b * H_ + h) * N_ + ns) * HD_ + dh] = f2bf(v);
                }
            }
        }
    }
}

// ---------------------------------------------------------------------------
// Flash attention: 128 query rows of one (b,h) per workgroup; K via TDM,
// V transposed manually into the next ping-pong buffer (overlapped).
// ---------------------------------------------------------------------------
__global__ __launch_bounds__(256)
void attn_kernel(const bf16_t* __restrict__ Q, const bf16_t* __restrict__ K,
                 const bf16_t* __restrict__ V, bf16_t* __restrict__ O)
{
    constexpr int BQ  = 128;
    constexpr int BKV = 64;
    constexpr int LDK = HD_ + 8;   // Ks [key][c]   272B rows
    constexpr int LDV = BKV + 8;   // Vs [c][key]   144B rows (transposed)
    constexpr int LDP = BKV + 8;   // per-wave P patch

    __shared__ bf16_t Ks[2][BKV * LDK];
    __shared__ bf16_t Vs[2][HD_ * LDV];
    __shared__ bf16_t Ps[8 * 16 * LDP];

    const int tid  = threadIdx.x;
    const int wave = tid >> 5;
    const int lane = tid & 31;
    const int l16  = lane & 15;
    const int lh   = lane >> 4;

    const int bh   = blockIdx.x / (N_ / BQ);
    const int qblk = blockIdx.x % (N_ / BQ);
    const int q0   = qblk * BQ + wave * 16;

    const bf16_t* Qh = Q + (size_t)bh * N_ * HD_;
    const bf16_t* Kh = K + (size_t)bh * N_ * HD_;
    const bf16_t* Vh = V + (size_t)bh * N_ * HD_;

    v16bf qf[4];
    {
        const bf16_t* qrow = Qh + (size_t)(q0 + l16) * HD_ + lh * 8;
        #pragma unroll
        for (int kt = 0; kt < 4; ++kt) {
            const bf16_t* p = qrow + kt * 32;
            qf[kt] = cat16(*(const v8bf*)p, *(const v8bf*)(p + 16));
        }
    }

    const v8f vzero = {0.f,0.f,0.f,0.f,0.f,0.f,0.f,0.f};
    v8f o[8];
    #pragma unroll
    for (int i = 0; i < 8; ++i) o[i] = vzero;
    float mj[8], lj[8];
    #pragma unroll
    for (int j = 0; j < 8; ++j) { mj[j] = -1e30f; lj[j] = 0.f; }

    bf16_t* Pw = Ps + wave * 16 * LDP;

    // V transpose fill (c-major reads are lane-coalesced; 4 keys packed per 8B)
    auto fill_v = [&](bf16_t* dst, int n0) {
        #pragma unroll
        for (int i = 0; i < 8; ++i) {
            int idx = tid + i * 256;
            int c   = idx & 127;
            int kq  = idx >> 7;
            const unsigned short* vp =
                (const unsigned short*)(Vh + (size_t)(n0 + kq * 4) * HD_ + c);
            unsigned w0 = (unsigned)vp[0] | ((unsigned)vp[HD_] << 16);
            unsigned w1 = (unsigned)vp[2 * HD_] | ((unsigned)vp[3 * HD_] << 16);
            *(uint2*)(dst + c * LDV + kq * 4) = make_uint2(w0, w1);
        }
    };

#if HAVE_TDM
    if (wave == 0) tdm_load_2d(&Ks[0][0], Kh, HD_, BKV, HD_, 5, 3);  // 256B+16B pad
#else
    fill_tile_64x128(&Ks[0][0], Kh, HD_, tid);
#endif
    fill_v(&Vs[0][0], 0);

    int cur = 0;
    for (int n0 = 0; n0 < N_; n0 += BKV) {
#if HAVE_TDM
        if (wave == 0) tdm_wait();
#endif
        __syncthreads();
        if (n0 + BKV < N_) {
#if HAVE_TDM
            if (wave == 0)
                tdm_load_2d(&Ks[cur ^ 1][0], Kh + (size_t)(n0 + BKV) * HD_,
                            HD_, BKV, HD_, 5, 3);
#else
            fill_tile_64x128(&Ks[cur ^ 1][0], Kh + (size_t)(n0 + BKV) * HD_, HD_, tid);
#endif
            fill_v(&Vs[cur ^ 1][0], n0 + BKV);
        }
        const bf16_t* Ksc = &Ks[cur][0];
        const bf16_t* Vsc = &Vs[cur][0];

        // S = Q * K^T : four 16-key tiles, contraction over HD (4 WMMAs each)
        v8f s[4];
        #pragma unroll
        for (int t = 0; t < 4; ++t) s[t] = vzero;
        #pragma unroll
        for (int kt = 0; kt < 4; ++kt) {
            v16bf kb[4];
            #pragma unroll
            for (int t = 0; t < 4; ++t)
                kb[t] = load_b_frag(Ksc, t * 16, LDK, kt * 32, l16, lh);
            #pragma unroll
            for (int t = 0; t < 4; ++t)
                s[t] = wmma_bf16(qf[kt], kb[t], s[t]);
        }

        // Online softmax (DPP 16-lane reductions; rows replicated per half)
        float alpha[8];
        #pragma unroll
        for (int j = 0; j < 8; ++j) {
            float mx = fmaxf(fmaxf(s[0][j], s[1][j]), fmaxf(s[2][j], s[3][j]));
            mx = red_max16(mx);
            const float mn = fmaxf(mj[j], mx);
            const float a  = __expf(mj[j] - mn);
            const float p0 = __expf(s[0][j] - mn);
            const float p1 = __expf(s[1][j] - mn);
            const float p2 = __expf(s[2][j] - mn);
            const float p3 = __expf(s[3][j] - mn);
            const float rs = red_sum16((p0 + p1) + (p2 + p3));
            lj[j] = lj[j] * a + rs;
            mj[j] = mn;
            alpha[j] = a;
            const int r = j + lh * 8;
            Pw[r * LDP +  0 + l16] = f2bf(p0);
            Pw[r * LDP + 16 + l16] = f2bf(p1);
            Pw[r * LDP + 32 + l16] = f2bf(p2);
            Pw[r * LDP + 48 + l16] = f2bf(p3);
        }
        #pragma unroll
        for (int nt = 0; nt < 8; ++nt)
            #pragma unroll
            for (int j = 0; j < 8; ++j)
                o[nt][j] *= alpha[j];

        // O += P * V
        #pragma unroll
        for (int ks = 0; ks < 2; ++ks) {
            const bf16_t* pp = Pw + l16 * LDP + ks * 32 + lh * 8;
            v16bf pf = cat16(*(const v8bf*)pp, *(const v8bf*)(pp + 16));
            #pragma unroll
            for (int nt = 0; nt < 8; ++nt) {
                v16bf vf = load_b_frag(Vsc, nt * 16, LDV, ks * 32, l16, lh);
                o[nt] = wmma_bf16(pf, vf, o[nt]);
            }
        }
        cur ^= 1;
    }

    // Normalize, store bf16 into [B*N, D] at column h*HD + c
    const int b = bh >> 4;
    const int h = bh & (H_ - 1);
    #pragma unroll
    for (int j = 0; j < 8; ++j) {
        const float rinv = 1.0f / lj[j];
        const int r = q0 + j + lh * 8;
        bf16_t* op = O + (size_t)(b * N_ + r) * D_ + h * HD_ + l16;
        #pragma unroll
        for (int nt = 0; nt < 8; ++nt)
            op[nt * 16] = f2bf(o[nt][j] * rinv);
    }
}

// ---------------------------------------------------------------------------
extern "C" void kernel_launch(void* const* d_in, const int* in_sizes, int n_in,
                              void* d_out, int out_size, void* d_ws, size_t ws_size,
                              hipStream_t stream) {
    (void)in_sizes; (void)n_in; (void)out_size; (void)ws_size;
    const float* x    = (const float*)d_in[0];
    const float* cosb = (const float*)d_in[1];
    const float* sinb = (const float*)d_in[2];
    const float* Wq   = (const float*)d_in[3];
    const float* bq   = (const float*)d_in[4];
    const float* Wk   = (const float*)d_in[5];
    const float* bk   = (const float*)d_in[6];
    const float* Wv   = (const float*)d_in[7];
    const float* bv   = (const float*)d_in[8];
    const float* Wo   = (const float*)d_in[9];
    const float* bo   = (const float*)d_in[10];
    float* out = (float*)d_out;

    char* ws = (char*)d_ws;
    const size_t eMD = (size_t)M_ * D_;     // 8.4M elems
    const size_t eDD = (size_t)D_ * D_;     // 4.2M elems
    bf16_t* xb  = (bf16_t*)ws;
    bf16_t* Wqt = xb  + eMD;
    bf16_t* Wkt = Wqt + eDD;
    bf16_t* Wvt = Wkt + eDD;
    bf16_t* Wot = Wvt + eDD;
    bf16_t* Qb  = Wot + eDD;
    bf16_t* Kb  = Qb  + eMD;
    bf16_t* Vb  = Kb  + eMD;
    bf16_t* Ob  = Vb  + eMD;

    dim3 block(256);
    // Pre-passes: x -> bf16; W -> transposed bf16
    cvt_x_kernel<<<dim3((unsigned)(eMD / (8 * 256))), block, 0, stream>>>(x, xb);
    dim3 tgrid(D_ / 64, D_ / 64);
    wtrans_kernel<<<tgrid, block, 0, stream>>>(Wq, Wqt);
    wtrans_kernel<<<tgrid, block, 0, stream>>>(Wk, Wkt);
    wtrans_kernel<<<tgrid, block, 0, stream>>>(Wv, Wvt);
    wtrans_kernel<<<tgrid, block, 0, stream>>>(Wo, Wot);

    dim3 ggrid(D_ / 128, M_ / 128);
    gemm_kernel<MODE_Q><<<ggrid, block, 0, stream>>>(xb, Wqt, bq, cosb, sinb, Qb, nullptr);
    gemm_kernel<MODE_K><<<ggrid, block, 0, stream>>>(xb, Wkt, bk, cosb, sinb, Kb, nullptr);
    gemm_kernel<MODE_V><<<ggrid, block, 0, stream>>>(xb, Wvt, bv, cosb, sinb, Vb, nullptr);

    attn_kernel<<<dim3(B_ * H_ * (N_ / 128)), block, 0, stream>>>(Qb, Kb, Vb, Ob);

    gemm_kernel<MODE_OUT><<<ggrid, block, 0, stream>>>(Ob, Wot, bo, nullptr, nullptr,
                                                       nullptr, out);
}
